// EfficientMemoryGEMM_67843303407871
// MI455X (gfx1250) — compile-verified, matrix operand order checked by reading, not run on hardware
//
#include <hip/hip_runtime.h>

typedef __attribute__((ext_vector_type(8))) int v8i;
typedef int v4ig __attribute__((__vector_size__(16)));   // matches builtin param

#define MDIM 8192
#define NDIM 8192
#define KDIM 2048
#define LDSS 80   // padded LDS row stride (bytes) for a 64-byte K-slice row

// ----- CDNA5 async global->LDS path (guarded so compile never breaks) ------
#if __has_builtin(__builtin_amdgcn_global_load_async_to_lds_b128)
#define HAVE_ASYNC_LDS 1
#define GL_ASYNC_B128(gsrc, ldst)                                             \
    __builtin_amdgcn_global_load_async_to_lds_b128(                           \
        (__attribute__((address_space(1))) v4ig*)(gsrc),                      \
        (__attribute__((address_space(3))) v4ig*)(ldst), 0, 0)
#if __has_builtin(__builtin_amdgcn_s_wait_asynccnt)
#define WAIT_ASYNC(n) __builtin_amdgcn_s_wait_asynccnt(n)
#else
#define WAIT_ASYNC(n) asm volatile("s_wait_asynccnt " #n)
#endif
#else
#define HAVE_ASYNC_LDS 0
#define WAIT_ASYNC(n)
#endif

// ---------------------------------------------------------------------------
// 0) zero the scale words + histograms (ws is poisoned with 0xAA by harness)
// ---------------------------------------------------------------------------
__global__ void init_kernel(unsigned int* ws) {
    int i = threadIdx.x;               // 512 threads: 2 scales + 255 + 255 bins
    if (i < 512) ws[i] = 0u;
}

// ---------------------------------------------------------------------------
// 1) per-tensor abs-max -> atomicMax on float bits (valid: values >= 0)
// ---------------------------------------------------------------------------
__global__ void absmax_kernel(const float* __restrict__ x, int nquads,
                              unsigned int* __restrict__ out) {
    __shared__ float red[256];
    float m = 0.0f;
    int stride = gridDim.x * blockDim.x;
    for (int i = blockIdx.x * blockDim.x + threadIdx.x; i < nquads; i += stride) {
        float4 v = ((const float4*)x)[i];
        m = fmaxf(m, fmaxf(fmaxf(fabsf(v.x), fabsf(v.y)),
                           fmaxf(fabsf(v.z), fabsf(v.w))));
    }
    red[threadIdx.x] = m;
    __syncthreads();
    for (int s = 128; s > 0; s >>= 1) {
        if (threadIdx.x < s)
            red[threadIdx.x] = fmaxf(red[threadIdx.x], red[threadIdx.x + s]);
        __syncthreads();
    }
    if (threadIdx.x == 0) atomicMax(out, __float_as_uint(red[0]));
}

// ---------------------------------------------------------------------------
// 2a) quantize x1 (row-major M x K) -> int8, accumulate 255-bin histogram
// ---------------------------------------------------------------------------
__device__ __forceinline__ int quant1(float v, float inv) {
    int q = (int)rintf(v * inv);
    q = q > 127 ? 127 : (q < -127 ? -127 : q);
    return q;
}

__global__ void quantA_kernel(const float* __restrict__ x,
                              const unsigned int* __restrict__ sbits,
                              signed char* __restrict__ q,
                              int* __restrict__ hist, int nquads) {
    __shared__ int h[255];
    for (int i = threadIdx.x; i < 255; i += blockDim.x) h[i] = 0;
    __syncthreads();
    const float inv = 127.0f / __uint_as_float(*sbits);
    int stride = gridDim.x * blockDim.x;
    for (int i = blockIdx.x * blockDim.x + threadIdx.x; i < nquads; i += stride) {
        float4 v = ((const float4*)x)[i];
        int q0 = quant1(v.x, inv), q1 = quant1(v.y, inv);
        int q2 = quant1(v.z, inv), q3 = quant1(v.w, inv);
        char4 c;
        c.x = (signed char)q0; c.y = (signed char)q1;
        c.z = (signed char)q2; c.w = (signed char)q3;
        ((char4*)q)[i] = c;
        atomicAdd(&h[q0 + 127], 1);
        atomicAdd(&h[q1 + 127], 1);
        atomicAdd(&h[q2 + 127], 1);
        atomicAdd(&h[q3 + 127], 1);
    }
    __syncthreads();
    for (int i = threadIdx.x; i < 255; i += blockDim.x)
        if (h[i]) atomicAdd(&hist[i], h[i]);
}

// ---------------------------------------------------------------------------
// 2b) quantize x2 (K x N) -> int8 TRANSPOSED (N x K) via LDS tile + histogram
// ---------------------------------------------------------------------------
__global__ void quantB_kernel(const float* __restrict__ x,
                              const unsigned int* __restrict__ sbits,
                              signed char* __restrict__ qT,
                              int* __restrict__ hist) {
    __shared__ signed char t[64 * 65];
    __shared__ int h[255];
    for (int i = threadIdx.x; i < 255; i += blockDim.x) h[i] = 0;
    __syncthreads();
    const float inv = 127.0f / __uint_as_float(*sbits);
    const int n0 = blockIdx.x * 64;   // along N (8192)
    const int k0 = blockIdx.y * 64;   // along K (2048)
    const int c  = threadIdx.x & 63;
    const int r0 = threadIdx.x >> 6;
    #pragma unroll
    for (int i = 0; i < 16; ++i) {
        int r = r0 + i * 4;                                   // k offset in tile
        float v = x[(size_t)(k0 + r) * NDIM + n0 + c];        // coalesced along N
        int qv = quant1(v, inv);
        t[c * 65 + r] = (signed char)qv;                      // transposed in LDS
        atomicAdd(&h[qv + 127], 1);
    }
    __syncthreads();
    #pragma unroll
    for (int i = 0; i < 16; ++i) {
        int r = r0 + i * 4;                                   // n offset in tile
        qT[(size_t)(n0 + r) * KDIM + k0 + c] = t[r * 65 + c]; // coalesced along K
    }
    __syncthreads();
    for (int i = threadIdx.x; i < 255; i += blockDim.x)
        if (h[i]) atomicAdd(&hist[i], h[i]);
}

// ---------------------------------------------------------------------------
// 3) int8 WMMA GEMM: C(8192x8192) in 128x128 tiles, 8 waves, wave = 32x64
//    Double-buffered LDS stages filled by GLOBAL_LOAD_ASYNC_TO_LDS_B128
//    (ASYNCcnt). Outer loop kept rolled (#pragma unroll 1) with two explicit
//    constant-stage phases to avoid full-unroll VGPR/address bloat.
// ---------------------------------------------------------------------------
__global__ void __launch_bounds__(256)
gemm_kernel(const signed char* __restrict__ qA,
            const signed char* __restrict__ qBT,
            const unsigned int* __restrict__ sbits,
            float* __restrict__ out) {
    __shared__ signed char As[2][128 * LDSS];
    __shared__ signed char Bs[2][128 * LDSS];

    const int tid  = threadIdx.x;
    const int lane = tid & 31;
    const int wave = tid >> 5;
    const int wm   = wave & 3;          // wave M offset = wm*32
    const int wn   = wave >> 2;         // wave N offset = wn*64
    const int m_blk = blockIdx.y * 128;
    const int n_blk = blockIdx.x * 128;

    const int half = lane >> 4;         // 0: K+0, 1: K+8  (8-bit operand layout)
    const int l15  = lane & 15;         // row (A) / col (B) within 16x16 tile

    // tile loader: 256 threads move 128x64 bytes per operand (32 B/thread)
    const int ldr_row = tid >> 1;
    const int ldr_col = (tid & 1) * 32;
    const size_t ga_off = (size_t)(m_blk + ldr_row) * KDIM + ldr_col;
    const size_t gb_off = (size_t)(n_blk + ldr_row) * KDIM + ldr_col;
    const int    l_off  = ldr_row * LDSS + ldr_col;

    v8i acc[2][4];
    #pragma unroll
    for (int mt = 0; mt < 2; ++mt)
        #pragma unroll
        for (int nt = 0; nt < 4; ++nt)
            acc[mt][nt] = (v8i){0, 0, 0, 0, 0, 0, 0, 0};

    // issue one K-slice fill (4 x 16B per thread) into LDS stage `st`
    auto issue_tile = [&](int st, int k0) {
        const signed char* ga = qA  + ga_off + k0;
        const signed char* gb = qBT + gb_off + k0;
        signed char* la = &As[st][l_off];
        signed char* lb = &Bs[st][l_off];
#if HAVE_ASYNC_LDS
        GL_ASYNC_B128(ga,      la);
        GL_ASYNC_B128(ga + 16, la + 16);
        GL_ASYNC_B128(gb,      lb);
        GL_ASYNC_B128(gb + 16, lb + 16);
#else
        int4 a0 = ((const int4*)ga)[0];
        int4 a1 = ((const int4*)ga)[1];
        int4 b0 = ((const int4*)gb)[0];
        int4 b1 = ((const int4*)gb)[1];
        *(int4*)(la)      = a0;
        *(int4*)(la + 16) = a1;
        *(int4*)(lb)      = b0;
        *(int4*)(lb + 16) = b1;
#endif
    };

    // 8 WMMAs against LDS stage `st` (st is a compile-time constant)
    auto compute_stage = [&](int st) {
        v8i afrag[2];
        #pragma unroll
        for (int mt = 0; mt < 2; ++mt) {
            const signed char* base =
                &As[st][(wm * 32 + mt * 16 + l15) * LDSS + half * 8];
            v8i f;
            #pragma unroll
            for (int g = 0; g < 4; ++g) {
                int2 d = *(const int2*)(base + g * 16);   // ds_load_b64
                f[2 * g]     = d.x;
                f[2 * g + 1] = d.y;
            }
            afrag[mt] = f;
        }
        v8i bfrag[4];
        #pragma unroll
        for (int nt = 0; nt < 4; ++nt) {
            const signed char* base =
                &Bs[st][(wn * 64 + nt * 16 + l15) * LDSS + half * 8];
            v8i f;
            #pragma unroll
            for (int g = 0; g < 4; ++g) {
                int2 d = *(const int2*)(base + g * 16);
                f[2 * g]     = d.x;
                f[2 * g + 1] = d.y;
            }
            bfrag[nt] = f;
        }
        #pragma unroll
        for (int mt = 0; mt < 2; ++mt)
            #pragma unroll
            for (int nt = 0; nt < 4; ++nt)
                acc[mt][nt] = __builtin_amdgcn_wmma_i32_16x16x64_iu8(
                    /*sgn_a=*/true, afrag[mt],
                    /*sgn_b=*/true, bfrag[nt],
                    acc[mt][nt], /*reuse_a=*/false, /*reuse_b=*/false);
    };

    issue_tile(0, 0);                    // prime stage 0

    #pragma unroll 1
    for (int k0 = 0; k0 < KDIM; k0 += 128) {
        // ---- phase A: fill stage 1 (k0+64), compute stage 0 (k0) ----
        __syncthreads();                 // all waves done reading stage 1
        issue_tile(1, k0 + 64);          // k0+64 <= 1984 < KDIM always
        WAIT_ASYNC(4);                   // in-order: stage-0 tile resident
        __syncthreads();

        if (k0 + 128 < KDIM) {           // warm L2 one outer trip ahead
            __builtin_prefetch(qA  + ga_off + k0 + 128, 0, 1);
            __builtin_prefetch(qBT + gb_off + k0 + 128, 0, 1);
        }
        compute_stage(0);

        // ---- phase B: fill stage 0 (k0+128), compute stage 1 (k0+64) ----
        __syncthreads();                 // all waves done reading stage 0
        if (k0 + 128 < KDIM) {
            issue_tile(0, k0 + 128);
            WAIT_ASYNC(4);               // stage-1 tile resident
        } else {
            WAIT_ASYNC(0);
        }
        __syncthreads();
        compute_stage(1);
    }

    const float s1 = __uint_as_float(sbits[0]);
    const float s2 = __uint_as_float(sbits[1]);
    const float scale = s1 * s2 / (127.0f * 127.0f);

    // C/D layout: VGPR j -> M = half*8 + j, N = l15
    #pragma unroll
    for (int mt = 0; mt < 2; ++mt) {
        #pragma unroll
        for (int nt = 0; nt < 4; ++nt) {
            const int col = n_blk + wn * 64 + nt * 16 + l15;
            #pragma unroll
            for (int j = 0; j < 8; ++j) {
                const int row = m_blk + wm * 32 + mt * 16 + half * 8 + j;
                out[(size_t)row * NDIM + col] = (float)acc[mt][nt][j] * scale;
            }
        }
    }
}

// ---------------------------------------------------------------------------
// 4) k-th smallest raw quantized value from the 255-bin histograms
//    k = int(16777216 * 0.9) = 15099494, 1-based -> first cumsum >= k
// ---------------------------------------------------------------------------
__global__ void kth_kernel(const int* __restrict__ hist1,
                           const int* __restrict__ hist2,
                           float* __restrict__ out) {
    if (threadIdx.x == 0) {
        const long long k = 15099494LL;
        long long c = 0; int val = 127;
        for (int i = 0; i < 255; ++i) {
            c += hist1[i];
            if (c >= k) { val = i - 127; break; }
        }
        out[(size_t)MDIM * NDIM] = (float)val;
        c = 0; val = 127;
        for (int i = 0; i < 255; ++i) {
            c += hist2[i];
            if (c >= k) { val = i - 127; break; }
        }
        out[(size_t)MDIM * NDIM + 1] = (float)val;
    }
}

// ---------------------------------------------------------------------------
extern "C" void kernel_launch(void* const* d_in, const int* in_sizes, int n_in,
                              void* d_out, int out_size, void* d_ws, size_t ws_size,
                              hipStream_t stream) {
    const float* x1 = (const float*)d_in[0];   // 8192 x 2048
    const float* x2 = (const float*)d_in[1];   // 2048 x 8192
    float* out = (float*)d_out;                // 8192*8192 result + 2 scalars

    // workspace layout
    unsigned int* sbits = (unsigned int*)d_ws;             // [0]=s1 bits, [1]=s2 bits
    int* hist1 = (int*)d_ws + 2;                           // 255 bins
    int* hist2 = (int*)d_ws + 257;                         // 255 bins (ends at 512 words)
    signed char* qA  = (signed char*)d_ws + 4096;          // 8192*2048 int8 (M x K)
    signed char* qBT = qA + (size_t)MDIM * KDIM;           // 8192*2048 int8 (N x K)

    const int nquads = (MDIM * KDIM) / 4;                  // 4,194,304

    init_kernel<<<1, 512, 0, stream>>>((unsigned int*)d_ws);
    absmax_kernel<<<2048, 256, 0, stream>>>(x1, nquads, &sbits[0]);
    absmax_kernel<<<2048, 256, 0, stream>>>(x2, nquads, &sbits[1]);
    quantA_kernel<<<4096, 256, 0, stream>>>(x1, &sbits[0], qA, hist1, nquads);
    quantB_kernel<<<dim3(NDIM / 64, KDIM / 64), 256, 0, stream>>>(x2, &sbits[1], qBT, hist2);
    gemm_kernel<<<dim3(NDIM / 128, MDIM / 128), 256, 0, stream>>>(qA, qBT, sbits, out);
    kth_kernel<<<1, 32, 0, stream>>>(hist1, hist2, out);
}
